// Net_66142496358824
// MI455X (gfx1250) — compile-verified, matrix operand order checked by reading, not run on hardware
//
#include <hip/hip_runtime.h>

typedef __attribute__((ext_vector_type(16))) _Float16 v16h;
typedef __attribute__((ext_vector_type(8)))  float    v8f;

#define Bn   2
#define Nn   8192
#define Kn   16
#define M1n  1639
#define M2n  410
#define NCLS 32
#define MAXNB 64

// ---------------------------------------------------------------- utilities
__device__ __forceinline__ float act_apply(float v, int act) {
  if (act == 1) return fmaxf(v, 0.0f);
  if (act == 2) return (v > 0.0f) ? v : 0.2f * v;
  return v;
}

__global__ void split_data(const float* __restrict__ data, float* __restrict__ coords,
                           float* __restrict__ feats, int total) {
  int t = blockIdx.x * blockDim.x + threadIdx.x;
  if (t >= total) return;
  const float* d = data + (size_t)t * 6;
  coords[t * 3 + 0] = d[0]; coords[t * 3 + 1] = d[1]; coords[t * 3 + 2] = d[2];
  feats[t * 3 + 0]  = d[3]; feats[t * 3 + 1]  = d[4]; feats[t * 3 + 2]  = d[5];
}

__global__ void copyf(const float* __restrict__ a, float* __restrict__ b, long n) {
  long i = (long)blockIdx.x * blockDim.x + threadIdx.x;
  if (i < n) b[i] = a[i];
}

// ------------------------------------------------------- naive gemm (tiny C)
__global__ void gemm_naive(const float* __restrict__ X, const float* __restrict__ W,
                           const float* __restrict__ bias, float* __restrict__ Y,
                           long rows, int K, int C, int act) {
  long t = (long)blockIdx.x * blockDim.x + threadIdx.x;
  if (t >= rows * (long)C) return;
  long r = t / C; int c = (int)(t % C);
  float s = bias ? bias[c] : 0.0f;
  const float* xr = X + r * K;
  for (int k = 0; k < K; ++k) s += xr[k] * W[(size_t)k * C + c];
  Y[t] = act_apply(s, act);
}

// ----------------------------------------------------- f16 staging for WMMA
__global__ void cvt_pad_f16(const float* __restrict__ src, _Float16* __restrict__ dst,
                            int rows, int K, int rowsP, int Kp) {
  long i = (long)blockIdx.x * blockDim.x + threadIdx.x;
  long total = (long)rowsP * Kp;
  if (i >= total) return;
  int r = (int)(i / Kp), k = (int)(i % Kp);
  float v = (r < rows && k < K) ? src[(size_t)r * K + k] : 0.0f;
  dst[i] = (_Float16)v;
}

__global__ void cvt_w_f16t(const float* __restrict__ W /*K x C*/,
                           _Float16* __restrict__ WT /*C x Kp*/, int K, int C, int Kp) {
  int i = blockIdx.x * blockDim.x + threadIdx.x;
  if (i >= C * Kp) return;
  int c = i / Kp, k = i % Kp;
  WT[i] = (_Float16)((k < K) ? W[(size_t)k * C + c] : 0.0f);
}

// ------------------------------------------------------------- WMMA GEMM
// NT 16x16 output tiles per wave32, register-blocked along N: one A-fragment
// feeds NT back-to-back v_wmma ops.  Explicitly double-buffered over K so the
// k+1 fragment loads are in flight while the k WMMA group executes; all NT B
// fragments are kept simultaneously live so the WMMA group issues without
// per-op loadcnt-0 stalls.  Fragment k order used here (A and B alike): lane l holds
// row/col (l&15), halves j -> K = (l>>4)*16 + j (contiguous 32B per lane).
template <int NT>
__global__ void wmma_gemm_t(const _Float16* __restrict__ A, const _Float16* __restrict__ BT,
                            const float* __restrict__ bias, float* __restrict__ Y,
                            int rows, int Kp, int C, int act, int tiles) {
  int wave = threadIdx.x >> 5;
  int lane = threadIdx.x & 31;
  int tile = blockIdx.x * 8 + wave;
  if (tile >= tiles) return;           // uniform per wave -> EXEC stays full
  int tilesN = (C >> 4) / NT;
  int tm = tile / tilesN, tn = tile - tm * tilesN;
  int arow = tm * 16 + (lane & 15);
  int khalf = (lane >> 4) << 4;
  const _Float16* ap = A + (size_t)arow * Kp + khalf;
  const _Float16* bp = BT + ((size_t)(tn * NT * 16) + (lane & 15)) * Kp + khalf;
  v8f acc[NT];
#pragma unroll
  for (int i = 0; i < NT; ++i) acc[i] = (v8f){};

  v16h av = *(const v16h*)ap;
  v16h bv[NT];
#pragma unroll
  for (int i = 0; i < NT; ++i) bv[i] = *(const v16h*)(bp + (size_t)i * 16 * Kp);

  for (int k = 32; k < Kp; k += 32) {
    // WGP-scope speculative prefetch 4 K-steps ahead (OOB is silently dropped)
    __builtin_prefetch(ap + k + 128, 0, 3);
    __builtin_prefetch(bp + k + 128, 0, 3);
    v16h avn = *(const v16h*)(ap + k);
    v16h bvn[NT];
#pragma unroll
    for (int i = 0; i < NT; ++i) bvn[i] = *(const v16h*)(bp + (size_t)i * 16 * Kp + k);
#pragma unroll
    for (int i = 0; i < NT; ++i)
      acc[i] = __builtin_amdgcn_wmma_f32_16x16x32_f16(false, av, false, bv[i],
                                                      (short)0, acc[i], false, false);
    av = avn;
#pragma unroll
    for (int i = 0; i < NT; ++i) bv[i] = bvn[i];
  }
#pragma unroll
  for (int i = 0; i < NT; ++i)
    acc[i] = __builtin_amdgcn_wmma_f32_16x16x32_f16(false, av, false, bv[i],
                                                    (short)0, acc[i], false, false);

  int rbase = tm * 16 + (lane >> 4) * 8;
#pragma unroll
  for (int i = 0; i < NT; ++i) {
    int bcol = tn * NT * 16 + i * 16 + (lane & 15);
    float bb = bias ? bias[bcol] : 0.0f;
#pragma unroll
    for (int v = 0; v < 8; ++v) {
      int r = rbase + v;
      if (r < rows) Y[(size_t)r * C + bcol] = act_apply(acc[i][v] + bb, act);
    }
  }
}

// ----------------------------------------------------------- batch norm
__global__ void bn_stats(const float* __restrict__ X, const unsigned char* __restrict__ mask,
                         float* __restrict__ stats, long rows, int C) {
  __shared__ float ss[256], s2[256], sc[256];
  int c = blockIdx.x;
  float a = 0.f, b = 0.f, n = 0.f;
  for (long r = threadIdx.x; r < rows; r += blockDim.x) {
    float m = mask ? (float)mask[r] : 1.0f;
    float v = X[r * C + c] * m;
    a += v; b += v * v; n += m;
  }
  ss[threadIdx.x] = a; s2[threadIdx.x] = b; sc[threadIdx.x] = n;
  __syncthreads();
  for (int st = 128; st > 0; st >>= 1) {
    if ((int)threadIdx.x < st) {
      ss[threadIdx.x] += ss[threadIdx.x + st];
      s2[threadIdx.x] += s2[threadIdx.x + st];
      sc[threadIdx.x] += sc[threadIdx.x + st];
    }
    __syncthreads();
  }
  if (threadIdx.x == 0) {
    float cnt = fmaxf(sc[0], 1.0f);
    float mean = ss[0] / cnt;
    float var = s2[0] / cnt - mean * mean;
    stats[c] = mean; stats[C + c] = fmaxf(var, 0.0f);
  }
}

__global__ void bn_apply(float* __restrict__ X, const float* __restrict__ stats,
                         const float* __restrict__ g, const float* __restrict__ beta,
                         long rows, int C, float eps, int act) {
  long i = (long)blockIdx.x * blockDim.x + threadIdx.x;
  if (i >= rows * (long)C) return;
  int c = (int)(i % C);
  float v = (X[i] - stats[c]) * rsqrtf(stats[C + c] + eps) * g[c] + beta[c];
  X[i] = act_apply(v, act);
}

// ----------------------------------------------------------- KNN-16 (self incl.)
__global__ void knn16(const float* __restrict__ coords, int* __restrict__ knn) {
  int t = blockIdx.x * blockDim.x + threadIdx.x;
  if (t >= Bn * Nn) return;
  int b = t / Nn, i = t % Nn;
  const float* cb = coords + (size_t)b * Nn * 3;
  float qx = cb[i * 3], qy = cb[i * 3 + 1], qz = cb[i * 3 + 2];
  float bd[Kn]; int bi[Kn];
  for (int j = 0; j < Kn; ++j) { bd[j] = INFINITY; bi[j] = 0; }
  for (int j = 0; j < Nn; ++j) {
    float dx = cb[j * 3] - qx, dy = cb[j * 3 + 1] - qy, dz = cb[j * 3 + 2] - qz;
    float d2 = dx * dx + dy * dy + dz * dz;
    if (d2 < bd[Kn - 1]) {
      int p = Kn - 1;
      while (p > 0 && d2 < bd[p - 1]) { bd[p] = bd[p - 1]; bi[p] = bi[p - 1]; --p; }
      bd[p] = d2; bi[p] = j;
    }
  }
  for (int j = 0; j < Kn; ++j) knn[(size_t)t * Kn + j] = bi[j];
}

__global__ void build_lse(const float* __restrict__ coords, const int* __restrict__ knn,
                          float* __restrict__ lse) {
  int t = blockIdx.x * blockDim.x + threadIdx.x;
  int total = Bn * Nn * Kn;
  if (t >= total) return;
  int n = (t / Kn) % Nn; int b = t / (Kn * Nn);
  const float* cb = coords + (size_t)b * Nn * 3;
  int j = knn[t];
  float cx = cb[n * 3], cy = cb[n * 3 + 1], cz = cb[n * 3 + 2];
  float nx = cb[j * 3], ny = cb[j * 3 + 1], nz = cb[j * 3 + 2];
  float dx = cx - nx, dy = cy - ny, dz = cz - nz;
  float* o = lse + (size_t)t * 10;
  o[0] = cx; o[1] = cy; o[2] = cz; o[3] = nx; o[4] = ny; o[5] = nz;
  o[6] = dx; o[7] = dy; o[8] = dz; o[9] = dx * dx + dy * dy + dz * dz;
}

__global__ void build_g(const float* __restrict__ h, const float* __restrict__ f,
                        float* __restrict__ g) {
  int t = blockIdx.x * blockDim.x + threadIdx.x;
  int total = Bn * Nn * Kn;
  if (t >= total) return;
  long pnt = t / Kn;   // b*N + n
  float* o = g + (size_t)t * 16;
  for (int c = 0; c < 8; ++c) o[c] = h[(size_t)t * 8 + c];
  for (int c = 0; c < 8; ++c) o[8 + c] = f[pnt * 8 + c];
}

// softmax over K per (b,n,channel), then pooled = sum_k softmax(s)*g
__global__ void softmax_pool(const float* __restrict__ s, const float* __restrict__ g,
                             float* __restrict__ pooled) {
  int t = blockIdx.x * blockDim.x + threadIdx.x;
  int total = Bn * Nn * 16;
  if (t >= total) return;
  int c = t % 16; long pnt = t / 16;
  float mx = -INFINITY;
  for (int k = 0; k < Kn; ++k) mx = fmaxf(mx, s[((size_t)pnt * Kn + k) * 16 + c]);
  float sum = 0.f;
  for (int k = 0; k < Kn; ++k) sum += expf(s[((size_t)pnt * Kn + k) * 16 + c] - mx);
  float acc = 0.f;
  for (int k = 0; k < Kn; ++k) {
    float w = expf(s[((size_t)pnt * Kn + k) * 16 + c] - mx) / sum;
    acc += w * g[((size_t)pnt * Kn + k) * 16 + c];
  }
  pooled[t] = acc;
}

// x0[b,n,c] = y[b, (n*8+c)%N, (n*8+c)/N]   (the reference's swapaxes+reshape)
__global__ void scramble_x0(const float* __restrict__ y, float* __restrict__ x0) {
  int t = blockIdx.x * blockDim.x + threadIdx.x;
  int total = Bn * Nn * 8;
  if (t >= total) return;
  int b = t / (Nn * 8); int r = t % (Nn * 8);
  x0[t] = y[(size_t)b * Nn * 8 + (size_t)(r % Nn) * 8 + (r / Nn)];
}

// --------------------------------------------------------------- FPS
__global__ void fps_kernel(const float* __restrict__ coords, int* __restrict__ out,
                           float* __restrict__ mind2, int Np, int M) {
  __shared__ float sv[256]; __shared__ int si[256];
  int b = blockIdx.x;
  const float* cb = coords + (size_t)b * Np * 3;
  int* ob = out + (size_t)b * M;
  float* md = mind2 + (size_t)b * Np;
  for (int i = threadIdx.x; i < Np; i += blockDim.x) md[i] = INFINITY;
  if (threadIdx.x == 0) ob[0] = 0;
  __syncthreads();
  int last = 0;
  for (int s = 1; s < M; ++s) {
    float lx = cb[last * 3], ly = cb[last * 3 + 1], lz = cb[last * 3 + 2];
    float best = -1.0f; int besti = 0;
    for (int i = threadIdx.x; i < Np; i += blockDim.x) {
      float dx = cb[i * 3] - lx, dy = cb[i * 3 + 1] - ly, dz = cb[i * 3 + 2] - lz;
      float m = fminf(md[i], dx * dx + dy * dy + dz * dz);
      md[i] = m;
      if (m > best) { best = m; besti = i; }
    }
    sv[threadIdx.x] = best; si[threadIdx.x] = besti;
    __syncthreads();
    for (int st = 128; st > 0; st >>= 1) {
      if ((int)threadIdx.x < st) {
        float v2 = sv[threadIdx.x + st]; int i2 = si[threadIdx.x + st];
        if (v2 > sv[threadIdx.x] || (v2 == sv[threadIdx.x] && i2 < si[threadIdx.x])) {
          sv[threadIdx.x] = v2; si[threadIdx.x] = i2;
        }
      }
      __syncthreads();
    }
    last = si[0];
    if (threadIdx.x == 0) ob[s] = last;
    __syncthreads();
  }
}

__global__ void gather3(const float* __restrict__ pos, const int* __restrict__ idx,
                        float* __restrict__ qpos, int Np, int Q) {
  int t = blockIdx.x * blockDim.x + threadIdx.x;
  if (t >= Bn * Q) return;
  int b = t / Q;
  int j = idx[t];
  const float* p = pos + ((size_t)b * Np + j) * 3;
  qpos[t * 3 + 0] = p[0]; qpos[t * 3 + 1] = p[1]; qpos[t * 3 + 2] = p[2];
}

// top-64 nearest within radius (set semantics; order irrelevant downstream)
__global__ void radius_topk(const float* __restrict__ pos, const float* __restrict__ qpos,
                            int* __restrict__ nidx, unsigned char* __restrict__ valid,
                            int Np, int Q, float r2) {
  int t = blockIdx.x * blockDim.x + threadIdx.x;
  if (t >= Bn * Q) return;
  int b = t / Q;
  const float* pb = pos + (size_t)b * Np * 3;
  float qx = qpos[t * 3], qy = qpos[t * 3 + 1], qz = qpos[t * 3 + 2];
  float d[MAXNB]; int id[MAXNB];
  int cnt = 0; float worst = -1.0f; int wi = 0;
  for (int j = 0; j < Np; ++j) {
    float dx = pb[j * 3] - qx, dy = pb[j * 3 + 1] - qy, dz = pb[j * 3 + 2] - qz;
    float d2 = dx * dx + dy * dy + dz * dz;
    if (d2 > r2) continue;
    if (cnt < MAXNB) {
      d[cnt] = d2; id[cnt] = j;
      if (d2 > worst) { worst = d2; wi = cnt; }
      ++cnt;
    } else if (d2 < worst) {
      d[wi] = d2; id[wi] = j;
      worst = -1.0f;
      for (int m = 0; m < MAXNB; ++m) if (d[m] > worst) { worst = d[m]; wi = m; }
    }
  }
  for (int m = 0; m < MAXNB; ++m) {
    nidx[(size_t)t * MAXNB + m]  = (m < cnt) ? id[m] : 0;
    valid[(size_t)t * MAXNB + m] = (m < cnt) ? 1 : 0;
  }
}

__global__ void build_pc_feat(const float* __restrict__ x, const float* __restrict__ pos,
                              const float* __restrict__ qpos, const int* __restrict__ nidx,
                              float* __restrict__ feat, int Np, int Q, int Cx) {
  int t = blockIdx.x * blockDim.x + threadIdx.x;
  int total = Bn * Q * MAXNB;
  if (t >= total) return;
  int q = (t / MAXNB) % Q; int b = t / (MAXNB * Q);
  int j = nidx[t];
  const float* xr = x + ((size_t)b * Np + j) * Cx;
  float* o = feat + (size_t)t * (Cx + 3);
  for (int c = 0; c < Cx; ++c) o[c] = xr[c];
  size_t pj = ((size_t)b * Np + j) * 3, pq = ((size_t)b * Q + q) * 3;
  o[Cx + 0] = pos[pj + 0] - qpos[pq + 0];
  o[Cx + 1] = pos[pj + 1] - qpos[pq + 1];
  o[Cx + 2] = pos[pj + 2] - qpos[pq + 2];
}

__global__ void masked_max_nb(const float* __restrict__ h, const unsigned char* __restrict__ valid,
                              float* __restrict__ out, int Q, int C) {
  long t = (long)blockIdx.x * blockDim.x + threadIdx.x;
  long total = (long)Bn * Q * C;
  if (t >= total) return;
  long bq = t / C; int c = (int)(t % C);
  float m = -INFINITY;
  for (int n = 0; n < MAXNB; ++n)
    if (valid[bq * MAXNB + n]) m = fmaxf(m, h[(bq * MAXNB + n) * (size_t)C + c]);
  out[t] = m;
}

__global__ void bmax_points(const float* __restrict__ X, float* __restrict__ out, int M, int C) {
  int t = blockIdx.x * blockDim.x + threadIdx.x;
  if (t >= Bn * C) return;
  int b = t / C, c = t % C;
  float m = -INFINITY;
  for (int j = 0; j < M; ++j) m = fmaxf(m, X[((size_t)b * M + j) * C + c]);
  out[t] = m;
}

__global__ void concat2(const float* __restrict__ A, const float* __restrict__ Bm,
                        float* __restrict__ O, long rows, int CA, int CB) {
  long i = (long)blockIdx.x * blockDim.x + threadIdx.x;
  int C = CA + CB;
  if (i >= rows * C) return;
  long r = i / C; int c = (int)(i % C);
  O[i] = (c < CA) ? A[r * CA + c] : Bm[r * CB + (c - CA)];
}

__global__ void concat_bcast(const float* __restrict__ x3, const float* __restrict__ x2,
                             float* __restrict__ O, int M, int C1, int C2) {
  long i = (long)blockIdx.x * blockDim.x + threadIdx.x;
  int C = C1 + C2; long total = (long)Bn * M * C;
  if (i >= total) return;
  long r = i / C; int c = (int)(i % C); int b = (int)(r / M);
  O[i] = (c < C1) ? x3[(size_t)b * C1 + c] : x2[r * C2 + (c - C1)];
}

__global__ void knn_interp3(const float* __restrict__ xs, const float* __restrict__ ps,
                            const float* __restrict__ pt, float* __restrict__ out,
                            int Ns, int Nt, int C) {
  int t = blockIdx.x * blockDim.x + threadIdx.x;
  if (t >= Bn * Nt) return;
  int b = t / Nt;
  float qx = pt[t * 3], qy = pt[t * 3 + 1], qz = pt[t * 3 + 2];
  const float* pb = ps + (size_t)b * Ns * 3;
  float d0 = INFINITY, d1 = INFINITY, d2 = INFINITY; int i0 = 0, i1 = 0, i2 = 0;
  for (int j = 0; j < Ns; ++j) {
    float dx = pb[j * 3] - qx, dy = pb[j * 3 + 1] - qy, dz = pb[j * 3 + 2] - qz;
    float dd = dx * dx + dy * dy + dz * dz;
    if (dd < d0)      { d2 = d1; i2 = i1; d1 = d0; i1 = i0; d0 = dd; i0 = j; }
    else if (dd < d1) { d2 = d1; i2 = i1; d1 = dd; i1 = j; }
    else if (dd < d2) { d2 = dd; i2 = j; }
  }
  float w0 = 1.0f / fmaxf(d0, 1e-16f), w1 = 1.0f / fmaxf(d1, 1e-16f), w2 = 1.0f / fmaxf(d2, 1e-16f);
  float wsum = w0 + w1 + w2;
  const float* a0 = xs + ((size_t)b * Ns + i0) * C;
  const float* a1 = xs + ((size_t)b * Ns + i1) * C;
  const float* a2 = xs + ((size_t)b * Ns + i2) * C;
  float* o = out + (size_t)t * C;
  for (int c = 0; c < C; ++c) o[c] = (a0[c] * w0 + a1[c] * w1 + a2[c] * w2) / wsum;
}

__global__ void logsoftmax32(const float* __restrict__ Z, float* __restrict__ out, long rows) {
  long r = (long)blockIdx.x * blockDim.x + threadIdx.x;
  if (r >= rows) return;
  const float* z = Z + r * NCLS;
  float mx = -INFINITY;
  for (int c = 0; c < NCLS; ++c) mx = fmaxf(mx, z[c]);
  float s = 0.f;
  for (int c = 0; c < NCLS; ++c) s += expf(z[c] - mx);
  float l = logf(s) + mx;
  for (int c = 0; c < NCLS; ++c) out[r * NCLS + c] = z[c] - l;
}

// ---------------------------------------------------------------- host side
static inline unsigned GD(long n, int b) { return (unsigned)((n + b - 1) / b); }

struct Bump {
  char* base; size_t off;
  void* take(size_t bytes) {
    off = (off + 255) & ~(size_t)255;
    void* p = base + off; off += bytes; return p;
  }
};

static void run_layer(const float* X, float* Y, const float* W, const float* bias,
                      int rows, int K, int C, int act,
                      _Float16* f16x, _Float16* f16w, hipStream_t s) {
  int Kp = (K + 31) & ~31;
  int rowsP = (rows + 15) & ~15;
  long tot = (long)rowsP * Kp;
  cvt_pad_f16<<<GD(tot, 256), 256, 0, s>>>(X, f16x, rows, K, rowsP, Kp);
  cvt_w_f16t<<<GD((long)C * Kp, 256), 256, 0, s>>>(W, f16w, K, C, Kp);
  if ((C & 63) == 0) {
    int tiles = (rowsP / 16) * (C / 64);
    wmma_gemm_t<4><<<GD(tiles, 8), 256, 0, s>>>(f16x, f16w, bias, Y, rows, Kp, C, act, tiles);
  } else {
    int tiles = (rowsP / 16) * (C / 16);
    wmma_gemm_t<1><<<GD(tiles, 8), 256, 0, s>>>(f16x, f16w, bias, Y, rows, Kp, C, act, tiles);
  }
}

static void run_bn(float* X, const float* g, const float* beta, const unsigned char* mask,
                   float* stats, long rows, int C, float eps, int act, hipStream_t s) {
  bn_stats<<<C, 256, 0, s>>>(X, mask, stats, rows, C);
  bn_apply<<<GD(rows * C, 256), 256, 0, s>>>(X, stats, g, beta, rows, C, eps, act);
}

// _mlp: per layer  x = relu(x@W+b); x = bn(x, eps=1e-5, mask)
static const float* run_mlp(void* const* d_in, int base, int nlayers, const int* dims,
                            float* bufA, float* bufB, long rows, const unsigned char* mask,
                            _Float16* f16x, _Float16* f16w, float* stats, hipStream_t s) {
  const float* in = bufA; float* out = bufB;
  for (int l = 0; l < nlayers; ++l) {
    const float* W  = (const float*)d_in[base + 4 * l + 0];
    const float* b  = (const float*)d_in[base + 4 * l + 1];
    const float* g  = (const float*)d_in[base + 4 * l + 2];
    const float* be = (const float*)d_in[base + 4 * l + 3];
    run_layer(in, out, W, b, (int)rows, dims[l], dims[l + 1], 1, f16x, f16w, s);
    run_bn(out, g, be, mask, stats, rows, dims[l + 1], 1e-5f, 0, s);
    const float* t = in; in = out; out = (float*)t;
  }
  return in;
}

extern "C" void kernel_launch(void* const* d_in, const int* in_sizes, int n_in,
                              void* d_out, int out_size, void* d_ws, size_t ws_size,
                              hipStream_t stream) {
  (void)in_sizes; (void)n_in; (void)out_size; (void)ws_size;
  const float* data = (const float*)d_in[0];
  // param indices (setup_inputs dict order)
  const int FC_W = 1, FC_B = 2, BN0_G = 3, BN0_B = 4, MLP1_W = 5, MLP1_B = 6;
  const int LSE_W = 7, LSE_B = 8, LSE_G = 9, LSE_BE = 10, POOL_WS = 11;
  const int POOL_W = 12, POOL_B = 13, POOL_G = 14, POOL_BE = 15;
  const int SA1 = 16, SA2 = 28, SA3 = 40, FP3 = 52, FP2 = 60, FP1 = 68;
  const int L1W = 80, L1B = 81, L2W = 82, L2B = 83, L3W = 84, L3B = 85;

  Bump ws{(char*)d_ws, 0};
  const long ARENA = 26853376;          // 209792 rows * 128 ch (largest stage)
  float* bufA = (float*)ws.take(sizeof(float) * ARENA);
  float* bufB = (float*)ws.take(sizeof(float) * ARENA);
  _Float16* f16x = (_Float16*)ws.take(sizeof(_Float16) * 13500000);
  _Float16* f16w = (_Float16*)ws.take(sizeof(_Float16) * 700000);
  float* stats  = (float*)ws.take(sizeof(float) * 2048);
  float* coords = (float*)ws.take(sizeof(float) * Bn * Nn * 3);
  float* feats  = (float*)ws.take(sizeof(float) * Bn * Nn * 3);
  float* xbuf   = (float*)ws.take(sizeof(float) * Bn * Nn * 8);
  float* fbuf   = (float*)ws.take(sizeof(float) * Bn * Nn * 8);
  float* pooled = (float*)ws.take(sizeof(float) * Bn * Nn * 16);
  float* ybuf   = (float*)ws.take(sizeof(float) * Bn * Nn * 8);
  float* x0     = (float*)ws.take(sizeof(float) * Bn * Nn * 8);
  int*   knn    = (int*)  ws.take(sizeof(int)   * Bn * Nn * Kn);
  float* mind2  = (float*)ws.take(sizeof(float) * Bn * Nn);
  int*   idx1   = (int*)  ws.take(sizeof(int)   * Bn * M1n);
  float* pos1   = (float*)ws.take(sizeof(float) * Bn * M1n * 3);
  int*   nidx1  = (int*)  ws.take(sizeof(int)   * Bn * M1n * MAXNB);
  unsigned char* val1 = (unsigned char*)ws.take(Bn * M1n * MAXNB);
  float* x1     = (float*)ws.take(sizeof(float) * Bn * M1n * 128);
  int*   idx2   = (int*)  ws.take(sizeof(int)   * Bn * M2n);
  float* pos2   = (float*)ws.take(sizeof(float) * Bn * M2n * 3);
  int*   nidx2  = (int*)  ws.take(sizeof(int)   * Bn * M2n * MAXNB);
  unsigned char* val2 = (unsigned char*)ws.take(Bn * M2n * MAXNB);
  float* x2     = (float*)ws.take(sizeof(float) * Bn * M2n * 256);
  float* x3     = (float*)ws.take(sizeof(float) * Bn * 1024);
  float* f3     = (float*)ws.take(sizeof(float) * Bn * M2n * 256);
  float* xi2    = (float*)ws.take(sizeof(float) * Bn * M1n * 256);
  float* f2     = (float*)ws.take(sizeof(float) * Bn * M1n * 128);
  float* xi1    = (float*)ws.take(sizeof(float) * Bn * Nn * 128);

  auto P = [&](int i) { return (const float*)d_in[i]; };

  // ---- stage 0: fc -> bn0 -> leaky ; knn ; mlp1 ; lse ; attention pool
  long BN_ = (long)Bn * Nn;
  split_data<<<GD(BN_, 256), 256, 0, stream>>>(data, coords, feats, (int)BN_);
  gemm_naive<<<GD(BN_ * 8, 256), 256, 0, stream>>>(feats, P(FC_W), P(FC_B), xbuf, BN_, 3, 8, 0);
  run_bn(xbuf, P(BN0_G), P(BN0_B), nullptr, stats, BN_, 8, 1e-6f, 2, stream);
  knn16<<<GD(BN_, 256), 256, 0, stream>>>(coords, knn);
  gemm_naive<<<GD(BN_ * 8, 256), 256, 0, stream>>>(xbuf, P(MLP1_W), P(MLP1_B), fbuf, BN_, 8, 8, 2);
  long BNK = BN_ * Kn;
  build_lse<<<GD(BNK, 256), 256, 0, stream>>>(coords, knn, bufA);
  gemm_naive<<<GD(BNK * 8, 256), 256, 0, stream>>>(bufA, P(LSE_W), P(LSE_B), bufB, BNK, 10, 8, 0);
  run_bn(bufB, P(LSE_G), P(LSE_BE), nullptr, stats, BNK, 8, 1e-6f, 1, stream);
  build_g<<<GD(BNK, 256), 256, 0, stream>>>(bufB, fbuf, bufA);       // g -> bufA (16ch)
  gemm_naive<<<GD(BNK * 16, 256), 256, 0, stream>>>(bufA, P(POOL_WS), nullptr, bufB, BNK, 16, 16, 0);
  softmax_pool<<<GD(BN_ * 16, 256), 256, 0, stream>>>(bufB, bufA, pooled);
  gemm_naive<<<GD(BN_ * 8, 256), 256, 0, stream>>>(pooled, P(POOL_W), P(POOL_B), ybuf, BN_, 16, 8, 0);
  run_bn(ybuf, P(POOL_G), P(POOL_BE), nullptr, stats, BN_, 8, 1e-6f, 1, stream);
  scramble_x0<<<GD(BN_ * 8, 256), 256, 0, stream>>>(ybuf, x0);

  // ---- SA1: fps -> radius(0.2) -> gather -> MLP(11,64,64,128) -> masked max
  fps_kernel<<<Bn, 256, 0, stream>>>(coords, idx1, mind2, Nn, M1n);
  gather3<<<GD(Bn * M1n, 256), 256, 0, stream>>>(coords, idx1, pos1, Nn, M1n);
  radius_topk<<<GD(Bn * M1n, 128), 128, 0, stream>>>(coords, pos1, nidx1, val1, Nn, M1n, 0.04f);
  long R1 = (long)Bn * M1n * MAXNB;
  build_pc_feat<<<GD(R1, 256), 256, 0, stream>>>(x0, coords, pos1, nidx1, bufA, Nn, M1n, 8);
  { const int dims[] = {11, 64, 64, 128};
    const float* h = run_mlp(d_in, SA1, 3, dims, bufA, bufB, R1, val1, f16x, f16w, stats, stream);
    masked_max_nb<<<GD((long)Bn * M1n * 128, 256), 256, 0, stream>>>(h, val1, x1, M1n, 128); }

  // ---- SA2: fps on pos1 -> radius(0.4) -> MLP(131,128,128,256) -> masked max
  fps_kernel<<<Bn, 256, 0, stream>>>(pos1, idx2, mind2, M1n, M2n);
  gather3<<<GD(Bn * M2n, 256), 256, 0, stream>>>(pos1, idx2, pos2, M1n, M2n);
  radius_topk<<<GD(Bn * M2n, 128), 128, 0, stream>>>(pos1, pos2, nidx2, val2, M1n, M2n, 0.16f);
  long R2 = (long)Bn * M2n * MAXNB;
  build_pc_feat<<<GD(R2, 256), 256, 0, stream>>>(x1, pos1, pos2, nidx2, bufA, M1n, M2n, 128);
  { const int dims[] = {131, 128, 128, 256};
    const float* h = run_mlp(d_in, SA2, 3, dims, bufA, bufB, R2, val2, f16x, f16w, stats, stream);
    masked_max_nb<<<GD((long)Bn * M2n * 256, 256), 256, 0, stream>>>(h, val2, x2, M2n, 256); }

  // ---- SA3 global: concat(x2,pos2) -> MLP(259,256,512,1024) -> max over points
  long RM2 = (long)Bn * M2n;
  concat2<<<GD(RM2 * 259, 256), 256, 0, stream>>>(x2, pos2, bufA, RM2, 256, 3);
  { const int dims[] = {259, 256, 512, 1024};
    const float* h = run_mlp(d_in, SA3, 3, dims, bufA, bufB, RM2, nullptr, f16x, f16w, stats, stream);
    bmax_points<<<GD(Bn * 1024, 256), 256, 0, stream>>>(h, x3, M2n, 1024); }

  // ---- FP3: concat(bcast x3, x2) -> MLP(1280,256,256)
  concat_bcast<<<GD(RM2 * 1280, 256), 256, 0, stream>>>(x3, x2, bufA, M2n, 1024, 256);
  { const int dims[] = {1280, 256, 256};
    const float* h = run_mlp(d_in, FP3, 2, dims, bufA, bufB, RM2, nullptr, f16x, f16w, stats, stream);
    copyf<<<GD(RM2 * 256, 256), 256, 0, stream>>>(h, f3, RM2 * 256); }

  // ---- FP2: interp(f3: pos2->pos1, k=3), concat x1 -> MLP(384,256,128)
  knn_interp3<<<GD(Bn * M1n, 128), 128, 0, stream>>>(f3, pos2, pos1, xi2, M2n, M1n, 256);
  long RM1 = (long)Bn * M1n;
  concat2<<<GD(RM1 * 384, 256), 256, 0, stream>>>(xi2, x1, bufA, RM1, 256, 128);
  { const int dims[] = {384, 256, 128};
    const float* h = run_mlp(d_in, FP2, 2, dims, bufA, bufB, RM1, nullptr, f16x, f16w, stats, stream);
    copyf<<<GD(RM1 * 128, 256), 256, 0, stream>>>(h, f2, RM1 * 128); }

  // ---- FP1: interp(f2: pos1->coords), concat x0 -> MLP(136,128,128,128)
  knn_interp3<<<GD(Bn * Nn, 128), 128, 0, stream>>>(f2, pos1, coords, xi1, M1n, Nn, 128);
  concat2<<<GD(BN_ * 136, 256), 256, 0, stream>>>(xi1, x0, bufA, BN_, 128, 8);
  const float* f1;
  { const int dims[] = {136, 128, 128, 128};
    f1 = run_mlp(d_in, FP1, 3, dims, bufA, bufB, BN_, nullptr, f16x, f16w, stats, stream); }

  // ---- head: lin1(relu) -> lin2 -> lin3 -> log_softmax
  float* hA = (f1 == bufA) ? bufB : bufA;
  float* hB = (f1 == bufA) ? bufA : bufB;
  run_layer(f1, hA, P(L1W), P(L1B), (int)BN_, 128, 128, 1, f16x, f16w, stream);
  run_layer(hA, hB, P(L2W), P(L2B), (int)BN_, 128, 64, 0, f16x, f16w, stream);
  run_layer(hB, hA, P(L3W), P(L3B), (int)BN_, 64, NCLS, 0, f16x, f16w, stream);
  logsoftmax32<<<GD(BN_, 256), 256, 0, stream>>>(hA, (float*)d_out, BN_);
}
